// PathSeqEncoder_46239617909501
// MI455X (gfx1250) — compile-verified, matrix operand order checked by reading, not run on hardware
//
#include <hip/hip_runtime.h>
#include <cstdint>
#include <cstddef>

// ---------------------------------------------------------------------------
// Types / helpers
// ---------------------------------------------------------------------------
typedef __bf16 bf16_t;
typedef __attribute__((ext_vector_type(16))) __bf16 v16bf;
typedef __attribute__((ext_vector_type(8)))  __bf16 v8bf;
typedef __attribute__((ext_vector_type(8)))  float  v8f;
typedef __attribute__((ext_vector_type(4)))  float  v4f;

#define NHEADS 8
#define DHEAD  32
#define HDIM   256
#define FFDIM  1024

__device__ __forceinline__ bf16_t f2bf(float f) { return (bf16_t)f; }   // native cvt
__device__ __forceinline__ float  bf2f(bf16_t b) { return (float)b; }

// A/B fragment for v_wmma_f32_16x16x32_bf16 (wave32).
// Per ISA layout: lane<16 holds row (lane&15), K = {0..7, 16..23}+k0;
// lane>=16 holds same row index, K = {8..15, 24..31}+k0. Caller passes the
// row pointer already offset by (k0 + (lane>>4)*8); we read +0..7 and +16..23.
__device__ __forceinline__ v16bf load_frag_bf16(const bf16_t* __restrict__ base) {
  v8bf lo = *(const v8bf*)(base);
  v8bf hi = *(const v8bf*)(base + 16);
  v16bf r;
#pragma unroll
  for (int i = 0; i < 8; ++i) { r[i] = lo[i]; r[8 + i] = hi[i]; }
  return r;
}

// ---------------------------------------------------------------------------
// Weight conversion f32 -> bf16 (8 elements per thread, vector loads/stores)
// ---------------------------------------------------------------------------
__global__ __launch_bounds__(256)
void cvt_bf16_kernel(const float* __restrict__ src, bf16_t* __restrict__ dst, int n8) {
  const int i = blockIdx.x * 256 + threadIdx.x;  // index over groups of 8
  if (i >= n8) return;
  const float* p = src + (size_t)i * 8;
  v4f a = *(const v4f*)p;
  v4f b = *(const v4f*)(p + 4);
  v8bf o;
#pragma unroll
  for (int j = 0; j < 4; ++j) { o[j] = f2bf(a[j]); o[4 + j] = f2bf(b[j]); }
  *(v8bf*)(dst + (size_t)i * 8) = o;
}

// ---------------------------------------------------------------------------
// Embedding + positional encoding + pad mask (dual f32 + bf16 output)
// ---------------------------------------------------------------------------
__device__ __forceinline__ float pos_enc(int l, int h) {
  const int i2 = h & ~1;
  const float dv  = __expf((float)i2 * (-9.210340371976184f / 256.0f)); // -ln(1e4)/H
  const float ang = (float)l * dv;
  return (h & 1) ? __cosf(ang) : __sinf(ang);
}

// Encoder 1: ctx[(dep*16+u), p] = units[u, (p/64)*32 + paths[dep,p]]
__global__ __launch_bounds__(256)
void embed_paths_kernel(const int* __restrict__ units, const int* __restrict__ paths,
                        const float* __restrict__ emb, float* __restrict__ x,
                        bf16_t* __restrict__ xb, unsigned char* __restrict__ mask) {
  const int t = blockIdx.x;                 // t = l*512 + p, l in [0,128)
  const int h = threadIdx.x;
  const int l = t >> 9;
  const int p = t & 511;
  const int dep = l >> 4, u = l & 15;
  const int col = (p >> 6) * 32 + paths[dep * 512 + p];
  const int tok = units[u * 256 + col];
  const float v = emb[(size_t)tok * HDIM + h] + pos_enc(l, h);
  x [(size_t)t * HDIM + h] = v;
  xb[(size_t)t * HDIM + h] = f2bf(v);
  if (h == 0) mask[p * 128 + l] = (tok == 0) ? 1 : 0;
}

// Encoder 2: order_tok[(upd*16+u), n] = units[u, n*32 + upd]
__global__ __launch_bounds__(256)
void embed_order_kernel(const int* __restrict__ units, const float* __restrict__ emb,
                        float* __restrict__ x, bf16_t* __restrict__ xb,
                        unsigned char* __restrict__ mask) {
  const int t = blockIdx.x;                 // t = i*8 + n, i in [0,512)
  const int h = threadIdx.x;
  const int i = t >> 3;
  const int n = t & 7;
  const int upd = i >> 4, u = i & 15;
  const int tok = units[u * 256 + n * 32 + upd];
  const float v = emb[(size_t)tok * HDIM + h] + pos_enc(i, h);
  x [(size_t)t * HDIM + h] = v;
  xb[(size_t)t * HDIM + h] = f2bf(v);
  if (h == 0) mask[n * 512 + i] = (tok == 0) ? 1 : 0;
}

// ---------------------------------------------------------------------------
// WMMA GEMM: Out[rows,N] = act(A[rows,K] @ W[N,K]^T + bias)
// A is bf16. Block = 256 threads = 8 waves; block tile 64(M) x 256(N);
// wave tile 32(M) x 64(N): per K-step 2 A-frags + 4 B-frags -> 8 WMMAs.
// ---------------------------------------------------------------------------
template<bool RELU, bool OUT_BF16>
__global__ __launch_bounds__(256)
void gemm_wmma_kernel(const bf16_t* __restrict__ A,
                      const bf16_t* __restrict__ W,
                      const float* __restrict__ bias,
                      void* __restrict__ Out,
                      int N, int K) {
  const int lane  = threadIdx.x & 31;
  const int wv    = threadIdx.x >> 5;
  const int mBase = blockIdx.y * 64 + (wv >> 2) * 32;
  const int nBase = blockIdx.x * 256 + (wv & 3) * 64;
  const int kSel  = (lane >> 4) * 8;
  const int mRow  = mBase + (lane & 15);

  v8f acc[2][4];
#pragma unroll
  for (int mi = 0; mi < 2; ++mi)
#pragma unroll
    for (int j = 0; j < 4; ++j) acc[mi][j] = (v8f){};

  const bf16_t* a0 = A + (size_t)mRow * K + kSel;
  const bf16_t* a1 = a0 + (size_t)16 * K;
  const bf16_t* wb = W + (size_t)(nBase + (lane & 15)) * K + kSel;

  for (int k0 = 0; k0 < K; k0 += 32) {
    v16bf fa0 = load_frag_bf16(a0 + k0);
    v16bf fa1 = load_frag_bf16(a1 + k0);
    if (k0 + 32 < K) {    // global_prefetch_b8 for the next K slab of A
      __builtin_prefetch(a0 + k0 + 32, 0, 1);
      __builtin_prefetch(a1 + k0 + 32, 0, 1);
    }
#pragma unroll
    for (int j = 0; j < 4; ++j) {
      v16bf fb = load_frag_bf16(wb + (size_t)j * 16 * K + k0);
      acc[0][j] = __builtin_amdgcn_wmma_f32_16x16x32_bf16(
          false, fa0, false, fb, (short)0, acc[0][j], false, false);
      acc[1][j] = __builtin_amdgcn_wmma_f32_16x16x32_bf16(
          false, fa1, false, fb, (short)0, acc[1][j], false, false);
    }
  }

  // C/D layout: VGPR r -> M = tileM + r + 8*(lane>=16), N = tileN + (lane&15)
  const int nCol0 = nBase + (lane & 15);
#pragma unroll
  for (int mi = 0; mi < 2; ++mi) {
    const int mOff = mBase + mi * 16 + ((lane >> 4) << 3);
#pragma unroll
    for (int j = 0; j < 4; ++j) {
      const int nCol = nCol0 + j * 16;
      const float bv = bias[nCol];
#pragma unroll
      for (int r = 0; r < 8; ++r) {
        const int row = mOff + r;
        float v = acc[mi][j][r] + bv;
        if (RELU) v = v > 0.0f ? v : 0.0f;
        if (OUT_BF16) ((bf16_t*)Out)[(size_t)row * N + nCol] = f2bf(v);
        else          ((float*) Out)[(size_t)row * N + nCol] = v;
      }
    }
  }
}

// ---------------------------------------------------------------------------
// Attention: one block (64 thr = 2 waves) handles one (b,h) and 2 query tiles.
// S-tile via WMMA; masked/scaled scores staged in LDS; softmax via shfl.
// qkv: [L*BN, 768] bf16 (q|k|v), out: [L*BN, 256] bf16
// ---------------------------------------------------------------------------
template<int L, int BN>
__global__ __launch_bounds__(64)
void attn_kernel(const bf16_t* __restrict__ qkv,
                 const unsigned char* __restrict__ mask,
                 bf16_t* __restrict__ out) {
  extern __shared__ float S_lds[];          // 2 waves * 16 rows * L floats
  const int tid   = threadIdx.x;
  const int lane  = tid & 31;
  const int wv    = tid >> 5;
  const int b     = blockIdx.x >> 3;
  const int h     = blockIdx.x & 7;
  const int qBase = (blockIdx.y * 2 + wv) * 16;
  float* S = S_lds + (size_t)wv * 16 * L;

  const int kSel = (lane >> 4) * 8;
  const int qRow = qBase + (lane & 15);
  const v16bf qa = load_frag_bf16(qkv + ((size_t)qRow * BN + b) * 768 + h * DHEAD + kSel);

  const float scale = 0.17677669529663687f; // 1/sqrt(32)
  for (int kt = 0; kt < L / 16; ++kt) {
    const int kRow = kt * 16 + (lane & 15);
    const v16bf kb = load_frag_bf16(
        qkv + ((size_t)kRow * BN + b) * 768 + 256 + h * DHEAD + kSel);
    v8f c = {};
    c = __builtin_amdgcn_wmma_f32_16x16x32_bf16(
        false, qa, false, kb, (short)0, c, false, false);
    const bool dead = mask[(size_t)b * L + kRow] != 0;
#pragma unroll
    for (int r = 0; r < 8; ++r) {
      const int m = r + ((lane >> 4) << 3);
      S[m * L + kRow] = dead ? -1.0e9f : c[r] * scale;
    }
  }

  // Row softmax stats: lane handles row (lane&15), halves split the key range.
  const int m    = lane & 15;
  const int half = lane >> 4;
  float mx = -1.0e30f;
  for (int k = half * (L / 2); k < (half + 1) * (L / 2); ++k)
    mx = fmaxf(mx, S[m * L + k]);
  mx = fmaxf(mx, __shfl_xor(mx, 16, 32));
  float sum = 0.0f;
  for (int k = half * (L / 2); k < (half + 1) * (L / 2); ++k)
    sum += __expf(S[m * L + k] - mx);
  sum += __shfl_xor(sum, 16, 32);
  const float inv = 1.0f / sum;

  // O = P @ V : lane owns (row m, 16 head dims starting at half*16)
  float o[16];
#pragma unroll
  for (int j = 0; j < 16; ++j) o[j] = 0.0f;
  for (int k = 0; k < L; ++k) {
    const float p = __expf(S[m * L + k] - mx) * inv;
    const bf16_t* vp = qkv + ((size_t)k * BN + b) * 768 + 512 + h * DHEAD + half * 16;
    v8bf v0 = *(const v8bf*)vp;
    v8bf v1 = *(const v8bf*)(vp + 8);
#pragma unroll
    for (int j = 0; j < 8; ++j) {
      o[j]     += p * bf2f(v0[j]);
      o[8 + j] += p * bf2f(v1[j]);
    }
  }
  const int orow = qBase + m;
  bf16_t* op = out + ((size_t)orow * BN + b) * HDIM + h * DHEAD + half * 16;
  v8bf o0, o1;
#pragma unroll
  for (int j = 0; j < 8; ++j) { o0[j] = f2bf(o[j]); o1[j] = f2bf(o[8 + j]); }
  *(v8bf*)op       = o0;
  *(v8bf*)(op + 8) = o1;
}

// ---------------------------------------------------------------------------
// Residual add + LayerNorm over H=256. 1 wave per row, fully coalesced.
// out = LN(in + res); optionally writes a bf16 shadow copy for the next GEMM.
// ---------------------------------------------------------------------------
template<bool DUAL>
__global__ __launch_bounds__(256)
void add_ln_kernel(const float* __restrict__ in, const float* __restrict__ res,
                   const float* __restrict__ gamma, const float* __restrict__ beta,
                   float* __restrict__ out, bf16_t* __restrict__ outb) {
  const int row  = blockIdx.x * 8 + (threadIdx.x >> 5);
  const int lane = threadIdx.x & 31;
  const float* p = in  + (size_t)row * HDIM + lane * 8;
  const float* q = res + (size_t)row * HDIM + lane * 8;
  float v[8];
  v4f a0 = *(const v4f*)p;
  v4f a1 = *(const v4f*)(p + 4);
  v4f b0 = *(const v4f*)q;
  v4f b1 = *(const v4f*)(q + 4);
#pragma unroll
  for (int i = 0; i < 4; ++i) { v[i] = a0[i] + b0[i]; v[4 + i] = a1[i] + b1[i]; }
  float s = 0.0f;
#pragma unroll
  for (int i = 0; i < 8; ++i) s += v[i];
#pragma unroll
  for (int o = 16; o >= 1; o >>= 1) s += __shfl_xor(s, o, 32);
  const float mean = s * (1.0f / 256.0f);
  float vs = 0.0f;
#pragma unroll
  for (int i = 0; i < 8; ++i) { const float d = v[i] - mean; vs += d * d; }
#pragma unroll
  for (int o = 16; o >= 1; o >>= 1) vs += __shfl_xor(vs, o, 32);
  const float rstd = rsqrtf(vs * (1.0f / 256.0f) + 1e-5f);
  const int col = lane * 8;
  v4f r0, r1;
#pragma unroll
  for (int i = 0; i < 4; ++i) {
    r0[i] = (v[i]     - mean) * rstd * gamma[col + i]     + beta[col + i];
    r1[i] = (v[4 + i] - mean) * rstd * gamma[col + 4 + i] + beta[col + 4 + i];
  }
  float* w = out + (size_t)row * HDIM + col;
  *(v4f*)w       = r0;
  *(v4f*)(w + 4) = r1;
  if (DUAL) {
    v8bf ob;
#pragma unroll
    for (int i = 0; i < 4; ++i) { ob[i] = f2bf(r0[i]); ob[4 + i] = f2bf(r1[i]); }
    *(v8bf*)(outb + (size_t)row * HDIM + col) = ob;
  }
}

// ---------------------------------------------------------------------------
// Host-side encoder pipeline
// ---------------------------------------------------------------------------
template<int L, int BN>
static void run_encoder(hipStream_t stream,
                        float* x, bf16_t* xb, bf16_t* qkvbuf, bf16_t* attnbuf,
                        float* rbuf, float* xln, bf16_t* xlnb, bf16_t* ffbuf,
                        bf16_t* const* wqkv, bf16_t* const* wout,
                        bf16_t* const* wff1, bf16_t* const* wff2,
                        const float* qkv_b, const float* out_b,
                        const float* ln1_s, const float* ln1_b,
                        const float* ff1_b, const float* ff2_b,
                        const float* ln2_s, const float* ln2_b,
                        const unsigned char* mask, float* finalOut) {
  const int T  = L * BN;
  const int CH = (T > 16384) ? 16384 : T;
  const size_t shmem = (size_t)2 * 16 * L * sizeof(float);
  for (int l = 0; l < 2; ++l) {
    // qkv = x @ qkv_w^T + b          (A bf16 -> out bf16)
    gemm_wmma_kernel<false, true><<<dim3(3, T / 64), 256, 0, stream>>>(
        xb, wqkv[l], qkv_b + (size_t)l * 768, qkvbuf, 768, 256);
    // attention
    attn_kernel<L, BN><<<dim3(BN * NHEADS, L / 32), 64, shmem, stream>>>(
        qkvbuf, mask, attnbuf);
    // o = attn @ out_w^T + b         (out f32; residual folded into LN)
    gemm_wmma_kernel<false, false><<<dim3(1, T / 64), 256, 0, stream>>>(
        attnbuf, wout[l], out_b + (size_t)l * 256, rbuf, 256, 256);
    add_ln_kernel<true><<<T / 8, 256, 0, stream>>>(
        rbuf, x, ln1_s + (size_t)l * 256, ln1_b + (size_t)l * 256, xln, xlnb);
    // FF in token chunks to bound the 1024-wide intermediate
    for (int c0 = 0; c0 < T; c0 += CH) {
      gemm_wmma_kernel<true, true><<<dim3(4, CH / 64), 256, 0, stream>>>(
          xlnb + (size_t)c0 * 256, wff1[l], ff1_b + (size_t)l * 1024,
          ffbuf, 1024, 256);
      gemm_wmma_kernel<false, false><<<dim3(1, CH / 64), 256, 0, stream>>>(
          ffbuf, wff2[l], ff2_b + (size_t)l * 256,
          rbuf + (size_t)c0 * 256, 256, 1024);
    }
    if (l == 1) {
      add_ln_kernel<false><<<T / 8, 256, 0, stream>>>(
          rbuf, xln, ln2_s + (size_t)l * 256, ln2_b + (size_t)l * 256,
          finalOut, nullptr);
    } else {
      add_ln_kernel<true><<<T / 8, 256, 0, stream>>>(
          rbuf, xln, ln2_s + (size_t)l * 256, ln2_b + (size_t)l * 256, x, xb);
    }
  }
}

// ---------------------------------------------------------------------------
// Entry point
// ---------------------------------------------------------------------------
extern "C" void kernel_launch(void* const* d_in, const int* in_sizes, int n_in,
                              void* d_out, int out_size, void* d_ws, size_t ws_size,
                              hipStream_t stream) {
  (void)in_sizes; (void)n_in; (void)out_size; (void)ws_size;
  const int*   units = (const int*)d_in[0];
  const int*   paths = (const int*)d_in[1];
  const float* emb   = (const float*)d_in[4];
  const float* EP[12];
  const float* OD[12];
  for (int i = 0; i < 12; ++i) {
    EP[i] = (const float*)d_in[5 + i];
    OD[i] = (const float*)d_in[17 + i];
  }

  const int T1 = 128 * 512;
  const int T2 = 512 * 8;

  size_t off = 0;
  auto alloc = [&](size_t bytes) -> void* {
    void* p = (void*)((char*)d_ws + off);
    off += (bytes + 255) & ~(size_t)255;
    return p;
  };

  // --- bf16 weights (converted every launch; deterministic) ---
  bf16_t* wqkv[2][2]; bf16_t* wout[2][2]; bf16_t* wff1[2][2]; bf16_t* wff2[2][2];
  for (int e = 0; e < 2; ++e) {
    const float* const* P = e ? OD : EP;
    for (int l = 0; l < 2; ++l) {
      wqkv[e][l] = (bf16_t*)alloc((size_t)196608 * 2);
      wout[e][l] = (bf16_t*)alloc((size_t)65536  * 2);
      wff1[e][l] = (bf16_t*)alloc((size_t)262144 * 2);
      wff2[e][l] = (bf16_t*)alloc((size_t)262144 * 2);
      cvt_bf16_kernel<<<196608 / 8 / 256, 256, 0, stream>>>(P[0] + (size_t)l * 196608, wqkv[e][l], 196608 / 8);
      cvt_bf16_kernel<<< 65536 / 8 / 256, 256, 0, stream>>>(P[2] + (size_t)l *  65536, wout[e][l],  65536 / 8);
      cvt_bf16_kernel<<<262144 / 8 / 256, 256, 0, stream>>>(P[6] + (size_t)l * 262144, wff1[e][l], 262144 / 8);
      cvt_bf16_kernel<<<262144 / 8 / 256, 256, 0, stream>>>(P[8] + (size_t)l * 262144, wff2[e][l], 262144 / 8);
    }
  }

  // --- activation buffers ---
  float*  x1    = (float*) alloc((size_t)T1 * HDIM * 4);
  bf16_t* x1b   = (bf16_t*)alloc((size_t)T1 * HDIM * 2);
  bf16_t* qkv1  = (bf16_t*)alloc((size_t)T1 * 768  * 2);
  bf16_t* att1  = (bf16_t*)alloc((size_t)T1 * HDIM * 2);
  float*  r1    = (float*) alloc((size_t)T1 * HDIM * 4);
  float*  xln1  = (float*) alloc((size_t)T1 * HDIM * 4);
  bf16_t* xln1b = (bf16_t*)alloc((size_t)T1 * HDIM * 2);
  bf16_t* ffb   = (bf16_t*)alloc((size_t)16384 * FFDIM * 2);
  unsigned char* mask1 = (unsigned char*)alloc(512 * 128);

  float*  x2    = (float*) alloc((size_t)T2 * HDIM * 4);
  bf16_t* x2b   = (bf16_t*)alloc((size_t)T2 * HDIM * 2);
  bf16_t* qkv2  = (bf16_t*)alloc((size_t)T2 * 768  * 2);
  bf16_t* att2  = (bf16_t*)alloc((size_t)T2 * HDIM * 2);
  float*  r2    = (float*) alloc((size_t)T2 * HDIM * 4);
  float*  xln2  = (float*) alloc((size_t)T2 * HDIM * 4);
  bf16_t* xln2b = (bf16_t*)alloc((size_t)T2 * HDIM * 2);
  unsigned char* mask2 = (unsigned char*)alloc(8 * 512);

  // --- embeddings + masks ---
  embed_paths_kernel<<<T1, 256, 0, stream>>>(units, paths, emb, x1, x1b, mask1);
  embed_order_kernel<<<T2, 256, 0, stream>>>(units, emb, x2, x2b, mask2);

  float* out1 = (float*)d_out;
  float* out2 = out1 + (size_t)T1 * HDIM;

  run_encoder<128, 512>(stream, x1, x1b, qkv1, att1, r1, xln1, xln1b, ffb,
                        wqkv[0], wout[0], wff1[0], wff2[0],
                        EP[1], EP[3], EP[4], EP[5], EP[7], EP[9], EP[10], EP[11],
                        mask1, out1);
  run_encoder<512, 8>(stream, x2, x2b, qkv2, att2, r2, xln2, xln2b, ffb,
                      wqkv[1], wout[1], wff1[1], wff2[1],
                      OD[1], OD[3], OD[4], OD[5], OD[7], OD[9], OD[10], OD[11],
                      mask2, out2);
}
